// MixtralBLockSparseTop2MLP_HQQ_6897717477959
// MI455X (gfx1250) — matrix operand order repaired and checked by reading, not compiled, not run on hardware
//
#include <hip/hip_runtime.h>
#include <hip/hip_bf16.h>

#define HIDDEN 4096
#define FFN    14336
#define G1     224     // FFN / 64
#define G2     64      // HIDDEN / 64
#define BM     128     // M rows per block
#define MT     8       // 16-row M tiles per wave

typedef __attribute__((ext_vector_type(16))) _Float16 v16h;
typedef __attribute__((ext_vector_type(8)))  float    v8f;

__device__ __forceinline__ v8f wmma_f16(v16h a, v16h b, v8f c) {
    return __builtin_amdgcn_wmma_f32_16x16x32_f16(
        /*neg_a=*/false, a, /*neg_b=*/false, b,
        /*c_mod=*/(short)0, c, /*reuse_a=*/false, /*reuse_b=*/false);
}

// CDNA5 async global->LDS copy, 16B per lane, tracked by ASYNCcnt.
// vdst = LDS byte address (VGPR), vaddr = 64-bit global address (VGPR pair).
__device__ __forceinline__ void async_ld_b128(unsigned lds_addr, const void* gaddr) {
    asm volatile("global_load_async_to_lds_b128 %0, %1, off"
                 :: "v"(lds_addr), "v"(gaddr) : "memory");
}
__device__ __forceinline__ void wait_asynccnt0() {
    asm volatile("s_wait_asynccnt 0x0" ::: "memory");
}
// Low 32 bits of a generic pointer to __shared__ == LDS byte offset.
__device__ __forceinline__ unsigned lds_off(const void* p) {
    return (unsigned)(uintptr_t)p;
}

// Dequantize 16 contiguous K-positions of one weight row (n fixed per lane).
// W[n,k] = ((q[k] >> shift) & 3 - z[g,k]) * s[g,k]
__device__ __forceinline__ v16h dq16(const int* __restrict__ qp,
                                     const float* __restrict__ sp,
                                     const float* __restrict__ zp,
                                     int shift) {
    union { int4   v4[4]; int   i[16]; } q;
    union { float4 v4[4]; float f[16]; } s, z;
#pragma unroll
    for (int j = 0; j < 4; ++j) {
        q.v4[j] = *(const int4*)  (qp + 4 * j);
        s.v4[j] = *(const float4*)(sp + 4 * j);
        z.v4[j] = *(const float4*)(zp + 4 * j);
    }
    union { v16h v; _Float16 h[16]; } r;
#pragma unroll
    for (int j = 0; j < 16; ++j) {
        float code = (float)((q.i[j] >> shift) & 3);
        r.h[j] = (_Float16)((code - z.f[j]) * s.f[j]);
    }
    return r.v;
}

// ---------------------------------------------------------------- x -> f16
__global__ void cvt_f32_to_f16(const float* __restrict__ x,
                               _Float16* __restrict__ y, int n) {
    int i = (blockIdx.x * blockDim.x + threadIdx.x) * 8;
    if (i + 7 < n) {
        float4 a = *(const float4*)(x + i);
        float4 b = *(const float4*)(x + i + 4);
        union { _Float16 h[8]; float4 v; } u;
        u.h[0] = (_Float16)a.x; u.h[1] = (_Float16)a.y;
        u.h[2] = (_Float16)a.z; u.h[3] = (_Float16)a.w;
        u.h[4] = (_Float16)b.x; u.h[5] = (_Float16)b.y;
        u.h[6] = (_Float16)b.z; u.h[7] = (_Float16)b.w;
        *(float4*)(y + i) = u.v;
    }
}

// -------------------------------------------------- fused gate/up + SwiGLU
// grid: (FFN/64, M/BM), block: 128 (4 waves). Double-buffered async LDS
// staging of the 128x32 f16 activation tile; one barrier per K-step.
__global__ void __launch_bounds__(128)
gateup_kernel(const _Float16* __restrict__ x16,
              const int*   __restrict__ qw1, const float* __restrict__ s1,
              const float* __restrict__ z1,
              const int*   __restrict__ qw3, const float* __restrict__ s3,
              const float* __restrict__ z3,
              _Float16* __restrict__ h16) {
    __shared__ __align__(16) _Float16 xs[2][BM][32];

    const int lane  = threadIdx.x & 31;
    const int wave  = threadIdx.x >> 5;
    const int ntile = blockIdx.x * 4 + wave;
    const int nn    = ntile * 16 + (lane & 15);   // output column (fixed/lane)
    const int mbase = blockIdx.y * BM;
    const int hi    = lane >> 4;                  // lane half select
    const int shift = (3 - (nn & 3)) * 2;
    const int g     = nn % G1;

    const int*   q1p = qw1 + (size_t)(nn >> 2) * HIDDEN;
    const int*   q3p = qw3 + (size_t)(nn >> 2) * HIDDEN;
    const float* s1p = s1  + (size_t)g * HIDDEN;
    const float* z1p = z1  + (size_t)g * HIDDEN;
    const float* s3p = s3  + (size_t)g * HIDDEN;
    const float* z3p = z3  + (size_t)g * HIDDEN;

    const _Float16* xrow = x16 + (size_t)(mbase + threadIdx.x) * HIDDEN;
    const unsigned  lbuf0 = lds_off(&xs[0][0][0]) + threadIdx.x * 64;
    const unsigned  lbuf1 = lds_off(&xs[1][0][0]) + threadIdx.x * 64;

    v8f accg[MT], accu[MT];
#pragma unroll
    for (int mt = 0; mt < MT; ++mt) { accg[mt] = (v8f){}; accu[mt] = (v8f){}; }

    // prologue: kick off async copy of first tile into buffer 0
#pragma unroll
    for (int j = 0; j < 4; ++j)
        async_ld_b128(lbuf0 + j * 16, (const char*)xrow + j * 16);

    int buf = 0;
    for (int kb = 0; kb < HIDDEN; kb += 32) {
        wait_asynccnt0();     // my stripe of buf landed in LDS
        __syncthreads();      // everyone's stripe landed; buf^1 fully consumed

        if (kb + 32 < HIDDEN) {   // overlap next tile copy with compute
            const unsigned  ldst = (buf ? lbuf0 : lbuf1);
            const char*     gsrc = (const char*)(xrow + kb + 32);
#pragma unroll
            for (int j = 0; j < 4; ++j)
                async_ld_b128(ldst + j * 16, gsrc + j * 16);
            __builtin_prefetch(q1p + kb + 32, 0, 0);
            __builtin_prefetch(q3p + kb + 32, 0, 0);
        }

        // B fragments (32x16: lanes 0-15 K=0..15, lanes 16-31 K=16..31)
        const int kB = kb + hi * 16;
        v16h b1 = dq16(q1p + kB, s1p + kB, z1p + kB, shift);
        v16h b3 = dq16(q3p + kB, s3p + kB, z3p + kB, shift);

        // A fragments from LDS (16x32: halves {c0..c0+15})
        const int r0 = lane & 15;
        const int c0 = hi * 16;
#pragma unroll
        for (int mt = 0; mt < MT; ++mt) {
            union { v16h v; float4 f[2]; } a;
            a.f[0] = *(const float4*)(&xs[buf][mt * 16 + r0][c0]);
            a.f[1] = *(const float4*)(&xs[buf][mt * 16 + r0][c0 + 8]);
            accg[mt] = wmma_f16(a.v, b1, accg[mt]);
            accu[mt] = wmma_f16(a.v, b3, accu[mt]);
        }
        buf ^= 1;
    }

    // epilogue: h = silu(gate) * up ; C layout lane<16: M=i, lane>=16: M=i+8
    const int mrow0 = mbase + hi * 8;
#pragma unroll
    for (int mt = 0; mt < MT; ++mt) {
#pragma unroll
        for (int i = 0; i < 8; ++i) {
            float gv = accg[mt][i];
            float uv = accu[mt][i];
            float hv = gv / (1.0f + __expf(-gv)) * uv;
            h16[(size_t)(mrow0 + mt * 16 + i) * FFN + nn] = (_Float16)hv;
        }
    }
}

// ------------------------------------------------------------- down proj
// out = h @ W2^T ; K = FFN, N = HIDDEN, fp32 output. Same async pipeline.
__global__ void __launch_bounds__(128)
down_kernel(const _Float16* __restrict__ h16,
            const int*   __restrict__ qw2, const float* __restrict__ s2,
            const float* __restrict__ z2,
            float* __restrict__ out) {
    __shared__ __align__(16) _Float16 hs[2][BM][32];

    const int lane  = threadIdx.x & 31;
    const int wave  = threadIdx.x >> 5;
    const int ntile = blockIdx.x * 4 + wave;
    const int nn    = ntile * 16 + (lane & 15);
    const int mbase = blockIdx.y * BM;
    const int hi    = lane >> 4;
    const int shift = (3 - (nn & 3)) * 2;
    const int g     = nn & (G2 - 1);

    const int*   qp = qw2 + (size_t)(nn >> 2) * FFN;
    const float* sp = s2  + (size_t)g * FFN;
    const float* zp = z2  + (size_t)g * FFN;

    const _Float16* hrow = h16 + (size_t)(mbase + threadIdx.x) * FFN;
    const unsigned  lbuf0 = lds_off(&hs[0][0][0]) + threadIdx.x * 64;
    const unsigned  lbuf1 = lds_off(&hs[1][0][0]) + threadIdx.x * 64;

    v8f acc[MT];
#pragma unroll
    for (int mt = 0; mt < MT; ++mt) acc[mt] = (v8f){};

#pragma unroll
    for (int j = 0; j < 4; ++j)
        async_ld_b128(lbuf0 + j * 16, (const char*)hrow + j * 16);

    int buf = 0;
    for (int kb = 0; kb < FFN; kb += 32) {
        wait_asynccnt0();
        __syncthreads();

        if (kb + 32 < FFN) {
            const unsigned  ldst = (buf ? lbuf0 : lbuf1);
            const char*     gsrc = (const char*)(hrow + kb + 32);
#pragma unroll
            for (int j = 0; j < 4; ++j)
                async_ld_b128(ldst + j * 16, gsrc + j * 16);
            __builtin_prefetch(qp + kb + 32, 0, 0);
        }

        const int kB = kb + hi * 16;
        v16h b = dq16(qp + kB, sp + kB, zp + kB, shift);

        const int r0 = lane & 15;
        const int c0 = hi * 16;
#pragma unroll
        for (int mt = 0; mt < MT; ++mt) {
            union { v16h v; float4 f[2]; } a;
            a.f[0] = *(const float4*)(&hs[buf][mt * 16 + r0][c0]);
            a.f[1] = *(const float4*)(&hs[buf][mt * 16 + r0][c0 + 8]);
            acc[mt] = wmma_f16(a.v, b, acc[mt]);
        }
        buf ^= 1;
    }

    const int mrow0 = mbase + hi * 8;
#pragma unroll
    for (int mt = 0; mt < MT; ++mt) {
#pragma unroll
        for (int i = 0; i < 8; ++i)
            out[(size_t)(mrow0 + mt * 16 + i) * HIDDEN + nn] = acc[mt][i];
    }
}

extern "C" void kernel_launch(void* const* d_in, const int* in_sizes, int n_in,
                              void* d_out, int out_size, void* d_ws, size_t ws_size,
                              hipStream_t stream) {
    const float* x   = (const float*)d_in[0];
    const int*   qw1 = (const int*)  d_in[1];
    const float* s1  = (const float*)d_in[2];
    const float* z1  = (const float*)d_in[3];
    const int*   qw3 = (const int*)  d_in[4];
    const float* s3  = (const float*)d_in[5];
    const float* z3  = (const float*)d_in[6];
    const int*   qw2 = (const int*)  d_in[7];
    const float* s2  = (const float*)d_in[8];
    const float* z2  = (const float*)d_in[9];
    float* out = (float*)d_out;

    const int M = in_sizes[0] / HIDDEN;   // 512

    _Float16* x16 = (_Float16*)d_ws;
    _Float16* h16 = x16 + (size_t)M * HIDDEN;

    // 1) x -> f16
    int nx = M * HIDDEN;
    cvt_f32_to_f16<<<nx / (256 * 8), 256, 0, stream>>>(x, x16, nx);

    // 2) fused gate/up + SwiGLU -> h16
    dim3 g1(FFN / 64, M / BM);
    gateup_kernel<<<g1, 128, 0, stream>>>(x16, qw1, s1, z1, qw3, s3, z3, h16);

    // 3) down projection -> fp32 out
    dim3 g2(HIDDEN / 64, M / BM);
    down_kernel<<<g2, 128, 0, stream>>>(h16, qw2, s2, z2, out);
}
